// Qwen3Attention_17849884082533
// MI455X (gfx1250) — matrix operand order repaired and checked by reading, not compiled, not run on hardware
//
#include <hip/hip_runtime.h>
#include <math.h>

// ---------------- types ----------------
typedef __bf16 bf16;
typedef __attribute__((ext_vector_type(16))) __bf16 bf16x16;
typedef __attribute__((ext_vector_type(8)))  __bf16 bf16x8;
typedef __attribute__((ext_vector_type(8)))  float  f32x8;

union FragU { bf16x16 v; bf16x8 h[2]; };

// ---------------- problem constants ----------------
#define BB     2
#define TT     2048
#define NH     32
#define NKV    8
#define HD     128
#define HID    4096
#define MTOK   (BB*TT)          // 4096 tokens
#define GROUPS (NH/NKV)         // 4

// ============================================================
// 1) fp32 -> bf16 convert
// ============================================================
__global__ __launch_bounds__(256)
void cvt_bf16_kernel(const float* __restrict__ in, bf16* __restrict__ out, size_t n) {
    size_t i = (size_t)blockIdx.x * blockDim.x + threadIdx.x;
    if (i < n) out[i] = (bf16)in[i];
}

// fp32 [K][N] row-major -> bf16 [N][K]  (so GEMM B-operand reads are contiguous in K)
__global__ __launch_bounds__(256)
void transpose_cvt_kernel(const float* __restrict__ in, bf16* __restrict__ out,
                          int Kd, int Nd) {
    size_t i = (size_t)blockIdx.x * blockDim.x + threadIdx.x;
    size_t total = (size_t)Kd * Nd;
    if (i >= total) return;
    int k = (int)(i / Nd);
    int n = (int)(i % Nd);
    out[(size_t)n * Kd + k] = (bf16)in[i];
}

// ============================================================
// 2) generic bf16 WMMA GEMM: C[M][N] = A[M][K] * BT[N][K]^T  (fp32 out)
//    block = 256 threads (8 waves, 2 M-waves x 4 N-waves), tile 128x128x32
// ============================================================
__global__ __launch_bounds__(256)
void gemm_bf16_kernel(const bf16* __restrict__ A, const bf16* __restrict__ BT,
                      float* __restrict__ C, int M, int N, int K) {
    __shared__ bf16 sA[128][32];
    __shared__ bf16 sB[128][32];

    const int tid  = threadIdx.x;
    const int lane = tid & 31, wave = tid >> 5;
    const int half = lane >> 4, l15 = lane & 15;
    const int wm = wave & 1, wn = wave >> 1;          // 2 x 4 wave grid
    const int m0 = blockIdx.y * 128, n0 = blockIdx.x * 128;
    const int ldr = tid >> 1;                          // load row 0..127
    const int seg = (tid & 1) * 16;                    // 0 or 16

    f32x8 acc[4][2] = {};

    for (int k0 = 0; k0 < K; k0 += 32) {
        *(bf16x16*)&sA[ldr][seg] = *(const bf16x16*)&A [(size_t)(m0 + ldr) * K + k0 + seg];
        *(bf16x16*)&sB[ldr][seg] = *(const bf16x16*)&BT[(size_t)(n0 + ldr) * K + k0 + seg];
        __syncthreads();

        bf16x16 bfrag[2];
#pragma unroll
        for (int nt = 0; nt < 2; ++nt)
            bfrag[nt] = *(const bf16x16*)&sB[wn * 32 + nt * 16 + l15][half * 16];

#pragma unroll
        for (int mt = 0; mt < 4; ++mt) {
            const bf16* ap = &sA[wm * 64 + mt * 16 + l15][half * 16];
            FragU a;
            a.h[0] = *(const bf16x8*)ap;          // K = half*16 + 0..7
            a.h[1] = *(const bf16x8*)(ap + 8);    // K = half*16 + 8..15
#pragma unroll
            for (int nt = 0; nt < 2; ++nt)
                acc[mt][nt] = __builtin_amdgcn_wmma_f32_16x16x32_bf16(
                    false, a.v, false, bfrag[nt], (short)0, acc[mt][nt], false, false);
        }
        __syncthreads();
    }

#pragma unroll
    for (int mt = 0; mt < 4; ++mt)
#pragma unroll
        for (int nt = 0; nt < 2; ++nt) {
            int n_g = n0 + wn * 32 + nt * 16 + l15;
            int m_b = m0 + wm * 64 + mt * 16 + half * 8;
#pragma unroll
            for (int e = 0; e < 8; ++e)
                C[(size_t)(m_b + e) * N + n_g] = acc[mt][nt][e];
        }
}

// ============================================================
// 3) fused RMSNorm + RoPE : raw fp32 [token][heads*128] -> bf16 [B][heads][T][128]
//    one block (128 threads) per (token, head)
// ============================================================
__global__ __launch_bounds__(128)
void norm_rope_kernel(const float* __restrict__ raw, const float* __restrict__ w,
                      bf16* __restrict__ out, int heads) {
    const int idx   = blockIdx.x;
    const int h     = idx % heads;
    const int token = idx / heads;
    const int t = token % TT, b = token / TT;
    const int d = threadIdx.x;

    __shared__ float red[4];
    __shared__ float xn[128];

    float v = raw[(size_t)token * (heads * HD) + h * HD + d];
    float ss = v * v;
#pragma unroll
    for (int off = 16; off >= 1; off >>= 1) ss += __shfl_xor(ss, off, 32);
    if ((threadIdx.x & 31) == 0) red[threadIdx.x >> 5] = ss;
    __syncthreads();
    float tot = red[0] + red[1] + red[2] + red[3];
    float inv = rsqrtf(tot * (1.0f / HD) + 1e-6f);
    xn[d] = v * inv * w[d];
    __syncthreads();

    int  j  = d & 63;
    bool hi = d >= 64;
    float timescale = __powf(1.0e6f, (2.0f * j) * (1.0f / HD));
    float ang = (float)t / timescale;
    float s = __sinf(ang), c = __cosf(ang);
    float a = xn[j], bv = xn[j + 64];
    float o = hi ? (bv * c + a * s) : (a * c - bv * s);
    out[(((size_t)(b * heads + h) * TT + t) * HD) + d] = (bf16)o;
}

// raw V fp32 [token][NKV*128] -> bf16 V^T [B][NKV][128][T]
__global__ __launch_bounds__(256)
void v_transpose_kernel(const float* __restrict__ vraw, bf16* __restrict__ vT) {
    size_t i = (size_t)blockIdx.x * blockDim.x + threadIdx.x;
    size_t total = (size_t)MTOK * NKV * HD;
    if (i >= total) return;
    int token = (int)(i / (NKV * HD));
    int r = (int)(i % (NKV * HD));
    int h = r / HD, d = r % HD;
    int b = token / TT, t = token % TT;
    vT[(((size_t)(b * NKV + h) * HD) + d) * TT + t] = (bf16)vraw[i];
}

// ============================================================
// 4) flash attention (causal, GQA 4:1), bf16 WMMA, fp32 online softmax
//    block = 256 threads handles (b, h, q-tile of 128)
// ============================================================
__global__ __launch_bounds__(256)
void attn_kernel(const bf16* __restrict__ Q, const bf16* __restrict__ Kmat,
                 const bf16* __restrict__ VT, bf16* __restrict__ Ob) {
    const int qt = blockIdx.x;
    const int bh = blockIdx.y;
    const int b = bh / NH, h = bh % NH, kvh = h / GROUPS;

    const bf16* Qh = Q    + ((size_t)(b * NH  + h)   * TT) * HD;   // [T][D]
    const bf16* Kh = Kmat + ((size_t)(b * NKV + kvh) * TT) * HD;   // [T][D]
    const bf16* Vh = VT   + ((size_t)(b * NKV + kvh) * HD) * TT;   // [D][T]

    __shared__ bf16 sA[128][32];          // Q d-chunk
    __shared__ bf16 sB[128][32];          // K d-chunk / V^T tk-chunk
    __shared__ bf16 sP[128][136];         // P staging (padded, keeps 16B align)
    __shared__ float sPM[4][128];
    __shared__ float sPS[4][128];
    __shared__ float sM[128], sL[128], sAl[128];

    const int tid  = threadIdx.x;
    const int lane = tid & 31, wave = tid >> 5;
    const int half = lane >> 4, l15 = lane & 15;
    const int wm = wave & 1, wn = wave >> 1;
    const int ldr = tid >> 1;
    const int seg = (tid & 1) * 16;
    const float scale = 0.08838834764831845f;   // 1/sqrt(128)

    f32x8 Oacc[4][2] = {};
    if (tid < 128) { sM[tid] = -3.0e38f; sL[tid] = 0.0f; }
    __syncthreads();

    for (int kb = 0; kb <= qt; ++kb) {
        // ---- S = (Q Kb^T) * scale ----
        f32x8 Sacc[4][2] = {};
#pragma unroll
        for (int ds = 0; ds < 4; ++ds) {
            *(bf16x16*)&sA[ldr][seg] = *(const bf16x16*)&Qh[(size_t)(qt * 128 + ldr) * HD + ds * 32 + seg];
            *(bf16x16*)&sB[ldr][seg] = *(const bf16x16*)&Kh[(size_t)(kb * 128 + ldr) * HD + ds * 32 + seg];
            __syncthreads();
            bf16x16 bfrag[2];
#pragma unroll
            for (int nt = 0; nt < 2; ++nt)
                bfrag[nt] = *(const bf16x16*)&sB[wn * 32 + nt * 16 + l15][half * 16];
#pragma unroll
            for (int mt = 0; mt < 4; ++mt) {
                const bf16* ap = &sA[wm * 64 + mt * 16 + l15][half * 16];
                FragU a;
                a.h[0] = *(const bf16x8*)ap;
                a.h[1] = *(const bf16x8*)(ap + 8);
#pragma unroll
                for (int nt = 0; nt < 2; ++nt)
                    Sacc[mt][nt] = __builtin_amdgcn_wmma_f32_16x16x32_bf16(
                        false, a.v, false, bfrag[nt], (short)0, Sacc[mt][nt], false, false);
            }
            __syncthreads();
        }

        const bool diag = (kb == qt);
#pragma unroll
        for (int mt = 0; mt < 4; ++mt)
#pragma unroll
            for (int nt = 0; nt < 2; ++nt) {
                int col = wn * 32 + nt * 16 + l15;
#pragma unroll
                for (int e = 0; e < 8; ++e) {
                    int row = wm * 64 + mt * 16 + half * 8 + e;
                    float s = Sacc[mt][nt][e] * scale;
                    if (diag && col > row) s = -3.0e38f;
                    Sacc[mt][nt][e] = s;
                }
            }

        // ---- partial row-max over this wave's 32 columns ----
#pragma unroll
        for (int mt = 0; mt < 4; ++mt)
#pragma unroll
            for (int e = 0; e < 8; ++e) {
                float pm = fmaxf(Sacc[mt][0][e], Sacc[mt][1][e]);
#pragma unroll
                for (int off = 1; off < 16; off <<= 1)
                    pm = fmaxf(pm, __shfl_xor(pm, off, 32));
                if (l15 == 0)
                    sPM[wn][wm * 64 + mt * 16 + half * 8 + e] = pm;
            }
        __syncthreads();

        if (tid < 128) {
            float nm = fmaxf(fmaxf(fmaxf(sPM[0][tid], sPM[1][tid]),
                                   fmaxf(sPM[2][tid], sPM[3][tid])), sM[tid]);
            sAl[tid] = __expf(sM[tid] - nm);
            sM[tid]  = nm;
        }
        __syncthreads();

        // ---- P = exp(S - m); rescale O; partial row-sums; stage P in LDS ----
#pragma unroll
        for (int mt = 0; mt < 4; ++mt)
#pragma unroll
            for (int e = 0; e < 8; ++e) {
                int row = wm * 64 + mt * 16 + half * 8 + e;
                float mrow = sM[row];
                float al   = sAl[row];
                float psum = 0.0f;
#pragma unroll
                for (int nt = 0; nt < 2; ++nt) {
                    int col = wn * 32 + nt * 16 + l15;
                    float p = __expf(Sacc[mt][nt][e] - mrow);
                    psum += p;
                    sP[row][col] = (bf16)p;
                    Oacc[mt][nt][e] *= al;
                }
#pragma unroll
                for (int off = 1; off < 16; off <<= 1)
                    psum += __shfl_xor(psum, off, 32);
                if (l15 == 0) sPS[wn][row] = psum;
            }
        __syncthreads();

        if (tid < 128)
            sL[tid] = sL[tid] * sAl[tid] + sPS[0][tid] + sPS[1][tid] + sPS[2][tid] + sPS[3][tid];

        // ---- O += P @ V  (contraction over 128 keys, 4 steps of 32) ----
#pragma unroll
        for (int ts = 0; ts < 4; ++ts) {
            *(bf16x16*)&sB[ldr][seg] =
                *(const bf16x16*)&Vh[(size_t)ldr * TT + kb * 128 + ts * 32 + seg];
            __syncthreads();
            bf16x16 bfrag[2];
#pragma unroll
            for (int nt = 0; nt < 2; ++nt)
                bfrag[nt] = *(const bf16x16*)&sB[wn * 32 + nt * 16 + l15][half * 16];
#pragma unroll
            for (int mt = 0; mt < 4; ++mt) {
                const bf16* pp = &sP[wm * 64 + mt * 16 + l15][ts * 32 + half * 16];
                FragU a;
                a.h[0] = *(const bf16x8*)pp;
                a.h[1] = *(const bf16x8*)(pp + 8);
#pragma unroll
                for (int nt = 0; nt < 2; ++nt)
                    Oacc[mt][nt] = __builtin_amdgcn_wmma_f32_16x16x32_bf16(
                        false, a.v, false, bfrag[nt], (short)0, Oacc[mt][nt], false, false);
            }
            __syncthreads();
        }
    }

    // ---- epilogue: O /= l ; store bf16 attn [b][t][h*128+d] ----
#pragma unroll
    for (int mt = 0; mt < 4; ++mt)
#pragma unroll
        for (int nt = 0; nt < 2; ++nt) {
            int col = wn * 32 + nt * 16 + l15;
#pragma unroll
            for (int e = 0; e < 8; ++e) {
                int row = wm * 64 + mt * 16 + half * 8 + e;
                float o = Oacc[mt][nt][e] / sL[row];
                size_t t = (size_t)qt * 128 + row;
                Ob[((size_t)b * TT + t) * (NH * HD) + h * HD + col] = (bf16)o;
            }
        }
}

// ============================================================
// host-side launcher
// ============================================================
extern "C" void kernel_launch(void* const* d_in, const int* in_sizes, int n_in,
                              void* d_out, int out_size, void* d_ws, size_t ws_size,
                              hipStream_t stream) {
    const float* x       = (const float*)d_in[0];
    const float* wq      = (const float*)d_in[1];
    const float* wk      = (const float*)d_in[2];
    const float* wv      = (const float*)d_in[3];
    const float* wo      = (const float*)d_in[4];
    const float* qnw     = (const float*)d_in[5];
    const float* knw     = (const float*)d_in[6];
    float*       out     = (float*)d_out;

    // workspace carve-up (256B aligned)
    char* w = (char*)d_ws;
    auto carve = [&](size_t bytes) {
        void* p = (void*)w;
        w += (bytes + 255) & ~(size_t)255;
        return p;
    };
    bf16*  xb    = (bf16*) carve((size_t)MTOK * HID * 2);
    bf16*  wqT   = (bf16*) carve((size_t)HID * (NH * HD) * 2);
    bf16*  wkT   = (bf16*) carve((size_t)HID * (NKV * HD) * 2);
    bf16*  wvT   = (bf16*) carve((size_t)HID * (NKV * HD) * 2);
    bf16*  woT   = (bf16*) carve((size_t)(NH * HD) * HID * 2);
    float* qraw  = (float*)carve((size_t)MTOK * (NH * HD) * 4);
    float* kraw  = (float*)carve((size_t)MTOK * (NKV * HD) * 4);
    float* vraw  = (float*)carve((size_t)MTOK * (NKV * HD) * 4);
    bf16*  qbuf  = (bf16*) carve((size_t)MTOK * (NH * HD) * 2);
    bf16*  kbuf  = (bf16*) carve((size_t)MTOK * (NKV * HD) * 2);
    bf16*  vTb   = (bf16*) carve((size_t)MTOK * (NKV * HD) * 2);
    bf16*  attnb = (bf16*) carve((size_t)MTOK * (NH * HD) * 2);

    // 1) conversions
    {
        size_t n = (size_t)MTOK * HID;
        cvt_bf16_kernel<<<(unsigned)((n + 255) / 256), 256, 0, stream>>>(x, xb, n);
    }
    {
        size_t n = (size_t)HID * (NH * HD);
        transpose_cvt_kernel<<<(unsigned)((n + 255) / 256), 256, 0, stream>>>(wq, wqT, HID, NH * HD);
    }
    {
        size_t n = (size_t)HID * (NKV * HD);
        transpose_cvt_kernel<<<(unsigned)((n + 255) / 256), 256, 0, stream>>>(wk, wkT, HID, NKV * HD);
        transpose_cvt_kernel<<<(unsigned)((n + 255) / 256), 256, 0, stream>>>(wv, wvT, HID, NKV * HD);
    }
    {
        size_t n = (size_t)(NH * HD) * HID;
        transpose_cvt_kernel<<<(unsigned)((n + 255) / 256), 256, 0, stream>>>(wo, woT, NH * HD, HID);
    }

    // 2) QKV projections (bf16 WMMA, fp32 out)
    gemm_bf16_kernel<<<dim3((NH * HD) / 128, MTOK / 128), 256, 0, stream>>>(xb, wqT, qraw, MTOK, NH * HD, HID);
    gemm_bf16_kernel<<<dim3((NKV * HD) / 128, MTOK / 128), 256, 0, stream>>>(xb, wkT, kraw, MTOK, NKV * HD, HID);
    gemm_bf16_kernel<<<dim3((NKV * HD) / 128, MTOK / 128), 256, 0, stream>>>(xb, wvT, vraw, MTOK, NKV * HD, HID);

    // 3) RMSNorm + RoPE -> bf16 (Q, K); V -> transposed bf16
    norm_rope_kernel<<<MTOK * NH, 128, 0, stream>>>(qraw, qnw, qbuf, NH);
    norm_rope_kernel<<<MTOK * NKV, 128, 0, stream>>>(kraw, knw, kbuf, NKV);
    {
        size_t n = (size_t)MTOK * NKV * HD;
        v_transpose_kernel<<<(unsigned)((n + 255) / 256), 256, 0, stream>>>(vraw, vTb);
    }

    // 4) causal flash attention
    attn_kernel<<<dim3(TT / 128, BB * NH), 256, 0, stream>>>(qbuf, kbuf, vTb, attnb);

    // 5) output projection -> fp32 d_out
    gemm_bf16_kernel<<<dim3(HID / 128, MTOK / 128), 256, 0, stream>>>(attnb, woT, out, MTOK, HID, NH * HD);
}
